// TransformerCrf_56882546868902
// MI455X (gfx1250) — matrix operand order repaired and checked by reading, not compiled, not run on hardware
//
#include <hip/hip_runtime.h>
#include <hip/hip_bf16.h>

// ---------------------------------------------------------------------------
// CDNA5 (gfx1250) wave32 WMMA transformer forward.
// bf16 operands + f32 accumulation via v_wmma_f32_16x16x32_bf16.
// Weights are pre-transposed/converted to bf16 [N,K] so every B-tile is a
// contiguous 32B run per lane (global_load_b128); activations carried as bf16
// copies written by fused epilogues. fp32 kept for residual/LN/softmax/logits.
// ---------------------------------------------------------------------------

typedef __attribute__((ext_vector_type(16))) __bf16 v16bf;
typedef __attribute__((ext_vector_type(8)))  float  v8f;

union ABu { v16bf v; unsigned int u[8]; };

// fp32 -> bf16 bits (round-half-up; 1-ulp tie diff vs RNE, negligible here)
__device__ __forceinline__ unsigned short bf_bits(float f) {
  return (unsigned short)((__float_as_uint(f) + 0x8000u) >> 16);
}
// pack two fp32 -> one dword of 2 bf16 using v_perm_b32
__device__ __forceinline__ unsigned int pack_bf2(float lo, float hi) {
  unsigned int a = __float_as_uint(lo) + 0x8000u;
  unsigned int b = __float_as_uint(hi) + 0x8000u;
  return __builtin_amdgcn_perm(b, a, 0x07060302u);  // {b[31:16], a[31:16]}
}

__device__ __forceinline__ v8f wmma_bf16(const ABu& a, const ABu& b, v8f c) {
  return __builtin_amdgcn_wmma_f32_16x16x32_bf16(
      false, a.v, false, b.v, (short)0, c, false, false);
}

#define BB   32
#define SS   512
#define DD   256
#define HH   8
#define HDIM 32
#define MFF  1024
#define LAB  20
#define NTOK (BB * SS)

typedef unsigned short ushort_t;

// ---------------------------------------------------------------------------
// Wt[n*K + k] = bf16(W[k*N + n])   (transpose + convert, tiny one-shot kernels)
// ---------------------------------------------------------------------------
__global__ void convert_wT_kernel(const float* __restrict__ W,
                                  ushort_t* __restrict__ Wt, int K, int N) {
  size_t i = (size_t)blockIdx.x * blockDim.x + threadIdx.x;
  if (i >= (size_t)K * N) return;
  int k = (int)(i % K);
  int n = (int)(i / K);
  Wt[i] = bf_bits(W[(size_t)k * N + n]);
}

// ---------------------------------------------------------------------------
// x = ch_emb[ids] + pos_emb ; also emit bf16 copy
// ---------------------------------------------------------------------------
__global__ void embed_kernel(const int* __restrict__ ids,
                             const float* __restrict__ ch_emb,
                             const float* __restrict__ pos_emb,
                             float* __restrict__ x,
                             ushort_t* __restrict__ xbf) {
  size_t idx = (size_t)blockIdx.x * blockDim.x + threadIdx.x;  // NTOK*DD
  int d   = (int)(idx % DD);
  int tok = (int)(idx / DD);
  int s   = tok % SS;
  float v = ch_emb[(size_t)ids[tok] * DD + d] + pos_emb[(size_t)s * DD + d];
  x[idx]   = v;
  xbf[idx] = bf_bits(v);
}

// ---------------------------------------------------------------------------
// Fast GEMM: C[M,N] = A[M,K] @ W[K,N] + bias  with A bf16 [M,K], Wt bf16 [N,K].
// One wave -> 16x64 output strip (4 WMMA acc tiles share one A tile).
// Requires M%16==0, N%64==0, K%32==0. Cf / Cbf outputs optional (null = skip).
// ---------------------------------------------------------------------------
__global__ void gemm_bf_fast(const ushort_t* __restrict__ Abf,
                             const ushort_t* __restrict__ Wt,
                             const float* __restrict__ bias,
                             float* __restrict__ Cf,
                             ushort_t* __restrict__ Cbf,
                             int Mrows, int Ncols, int Kdim) {
  const int lane = threadIdx.x & 31;
  const int wave = threadIdx.x >> 5;
  const int half = lane >> 4;
  const int l15  = lane & 15;

  const int nstrips = Ncols >> 6;
  const int tile = blockIdx.x * 8 + wave;          // uniform per wave
  if (tile >= (Mrows >> 4) * nstrips) return;
  const int mt = tile / nstrips;
  const int nt = tile % nstrips;
  const int m0 = mt << 4, n0 = nt << 6;

  const ushort_t* Ar = Abf + (size_t)(m0 + l15) * Kdim;

  v8f acc0 = {}, acc1 = {}, acc2 = {}, acc3 = {};
  for (int kk = 0; kk < Kdim; kk += 32) {
    ABu a;
#pragma unroll
    for (int i = 0; i < 8; ++i) {
      int k0 = (half << 4) + (i << 1);
      a.u[i] = *(const unsigned int*)(Ar + kk + k0);   // 2x b128 per lane
    }
#pragma unroll
    for (int sub = 0; sub < 4; ++sub) {
      const ushort_t* Br =
          Wt + (size_t)(n0 + sub * 16 + l15) * Kdim + kk + (half << 4);
      ABu b;
#pragma unroll
      for (int j = 0; j < 8; ++j)
        b.u[j] = *(const unsigned int*)(Br + 2 * j);   // 2x b128 per lane
      v8f& acc = sub == 0 ? acc0 : sub == 1 ? acc1 : sub == 2 ? acc2 : acc3;
      acc = wmma_bf16(a, b, acc);
    }
  }

#pragma unroll
  for (int sub = 0; sub < 4; ++sub) {
    const v8f& acc = sub == 0 ? acc0 : sub == 1 ? acc1 : sub == 2 ? acc2 : acc3;
    const int bcol = n0 + sub * 16 + l15;
    const float bv = bias[bcol];
#pragma unroll
    for (int r = 0; r < 8; ++r) {
      float val = acc[r] + bv;
      size_t off = (size_t)(m0 + r + (half << 3)) * Ncols + bcol;
      if (Cf)  Cf[off]  = val;        // uniform branch
      if (Cbf) Cbf[off] = bf_bits(val);
    }
  }
}

// ---------------------------------------------------------------------------
// Guarded GEMM (final logits, N=20): 16x16 per wave, clamped B addresses +
// cndmask zero-fill (no exec-masked load sequences).
// ---------------------------------------------------------------------------
__global__ void gemm_bf_guard(const ushort_t* __restrict__ Abf,
                              const ushort_t* __restrict__ Wt,
                              const float* __restrict__ bias,
                              float* __restrict__ Cf,
                              int Mrows, int Ncols, int Kdim) {
  const int lane = threadIdx.x & 31;
  const int wave = threadIdx.x >> 5;
  const int half = lane >> 4;
  const int l15  = lane & 15;

  const int ntiles = (Ncols + 15) >> 4;
  const int tile = blockIdx.x * 8 + wave;
  if (tile >= (Mrows >> 4) * ntiles) return;
  const int mt = tile / ntiles;
  const int nt = tile % ntiles;
  const int m0 = mt << 4, n0 = nt << 4;

  const int  bcol = n0 + l15;
  const bool bok  = bcol < Ncols;
  const int  brow = bok ? bcol : (Ncols - 1);     // clamp: loads always legal

  const ushort_t* Ar = Abf + (size_t)(m0 + l15) * Kdim;
  const ushort_t* Br = Wt + (size_t)brow * Kdim;

  v8f acc = {};
  for (int kk = 0; kk < Kdim; kk += 32) {
    ABu a, b;
#pragma unroll
    for (int i = 0; i < 8; ++i) {
      int k0 = (half << 4) + (i << 1);
      a.u[i] = *(const unsigned int*)(Ar + kk + k0);
    }
#pragma unroll
    for (int j = 0; j < 8; ++j) {
      unsigned int raw = *(const unsigned int*)(Br + kk + (half << 4) + 2 * j);
      b.u[j] = bok ? raw : 0u;                    // v_cndmask, not exec dance
    }
    acc = wmma_bf16(a, b, acc);
  }

  if (bok) {
    const float bv = bias[bcol];
#pragma unroll
    for (int r = 0; r < 8; ++r)
      Cf[(size_t)(m0 + r + (half << 3)) * Ncols + bcol] = acc[r] + bv;
  }
}

// ---------------------------------------------------------------------------
// Attention: one wave per (b, h, 16-query tile). Q/K/V are bf16 [tok, D].
// scores = QK^T/16 (one WMMA per key tile; head_dim==K step==32),
// key-mask -> fp32 softmax in LDS -> query-mask -> P@V (2x16 WMMAs).
// ---------------------------------------------------------------------------
__global__ void attention_kernel(const ushort_t* __restrict__ Qbf,
                                 const ushort_t* __restrict__ Kbf,
                                 const ushort_t* __restrict__ Vbf,
                                 const int* __restrict__ slen,
                                 float* __restrict__ Out) {
  __shared__ float sc[16][SS];   // 32 KB of 320 KB/WGP

  const int lane = threadIdx.x & 31;
  const int half = lane >> 4;
  const int l15  = lane & 15;

  const int qt = blockIdx.x, h = blockIdx.y, b = blockIdx.z;
  const int q0 = qt * 16;
  const int len = slen[b];

  // A tile: Q rows q0..q0+15, full head_dim (reused across all key tiles)
  ABu aq;
  {
    const ushort_t* Aq = Qbf + ((size_t)b * SS + q0 + l15) * DD + h * HDIM;
#pragma unroll
    for (int i = 0; i < 8; ++i) {
      int k0 = (half << 4) + (i << 1);
      aq.u[i] = *(const unsigned int*)(Aq + k0);
    }
  }
  const float invs = 0.0625f;                // 1/sqrt(256)
  const float NEGF = -4.2949673e9f;          // -(2^32-1)

  for (int kt = 0; kt < SS / 16; ++kt) {
    const int key = kt * 16 + l15;
    const ushort_t* Kp =
        Kbf + ((size_t)b * SS + key) * DD + h * HDIM + (half << 4);
    ABu bk;
#pragma unroll
    for (int j = 0; j < 8; ++j) bk.u[j] = *(const unsigned int*)(Kp + 2 * j);

    v8f c = {};
    c = wmma_bf16(aq, bk, c);
#pragma unroll
    for (int r = 0; r < 8; ++r) {
      float s = c[r] * invs;
      if (key >= len) s = NEGF;              // key mask
      sc[r + (half << 3)][kt * 16 + l15] = s;
    }
  }
  __syncthreads();

  // fp32 softmax per row, 32 lanes cooperating
  for (int m = 0; m < 16; ++m) {
    float mx = -3.4e38f;
    for (int j = lane; j < SS; j += 32) mx = fmaxf(mx, sc[m][j]);
#pragma unroll
    for (int o = 16; o; o >>= 1) mx = fmaxf(mx, __shfl_xor(mx, o, 32));
    float sum = 0.0f;
    for (int j = lane; j < SS; j += 32) {
      float e = __expf(sc[m][j] - mx);
      sc[m][j] = e;
      sum += e;
    }
#pragma unroll
    for (int o = 16; o; o >>= 1) sum += __shfl_xor(sum, o, 32);
    float inv = (q0 + m < len) ? (1.0f / sum) : 0.0f;  // query mask folded in
    for (int j = lane; j < SS; j += 32) sc[m][j] *= inv;
  }
  __syncthreads();

  // out = P @ V : N = head_dim = 32 -> two 16-col tiles, K = 512
#pragma unroll
  for (int nt = 0; nt < 2; ++nt) {
    v8f acc = {};
    for (int kt = 0; kt < SS / 32; ++kt) {
      const int kk = kt * 32;
      ABu ap, bv;
#pragma unroll
      for (int i = 0; i < 8; ++i) {
        int k0 = (half << 4) + (i << 1);
        ap.u[i] = pack_bf2(sc[l15][kk + k0], sc[l15][kk + k0 + 1]);
      }
      const ushort_t* Vp = Vbf + ((size_t)b * SS + kk + (half << 4)) * DD +
                           h * HDIM + nt * 16 + l15;
#pragma unroll
      for (int j = 0; j < 8; ++j) {
        unsigned int lo = Vp[(size_t)(2 * j) * DD];
        unsigned int hi = Vp[(size_t)(2 * j + 1) * DD];
        bv.u[j] = lo | (hi << 16);
      }
      acc = wmma_bf16(ap, bv, acc);
    }
#pragma unroll
    for (int r = 0; r < 8; ++r)
      Out[((size_t)b * SS + q0 + r + (half << 3)) * DD + h * HDIM + nt * 16 +
          l15] = acc[r];
  }
}

// ---------------------------------------------------------------------------
// x = LN(t + x) (biased var, eps=1e-23 inside sqrt); also emit bf16 copy of x
// ---------------------------------------------------------------------------
__global__ void add_ln_kernel(const float* __restrict__ t,
                              float* __restrict__ x,
                              ushort_t* __restrict__ xbf,
                              const float* __restrict__ gamma,
                              const float* __restrict__ beta) {
  __shared__ float red[DD];
  const int row = blockIdx.x;
  const int d   = threadIdx.x;
  const size_t idx = (size_t)row * DD + d;

  float v = t[idx] + x[idx];
  red[d] = v;
  __syncthreads();
#pragma unroll
  for (int o = 128; o; o >>= 1) {
    if (d < o) red[d] += red[d + o];
    __syncthreads();
  }
  const float mean = red[0] * (1.0f / DD);
  __syncthreads();
  const float c = v - mean;
  red[d] = c * c;
  __syncthreads();
#pragma unroll
  for (int o = 128; o; o >>= 1) {
    if (d < o) red[d] += red[d + o];
    __syncthreads();
  }
  const float var = red[0] * (1.0f / DD);

  float y = gamma[d] * c / sqrtf(var + 1e-23f) + beta[d];
  x[idx]   = y;
  xbf[idx] = bf_bits(y);
}

// ---------------------------------------------------------------------------
extern "C" void kernel_launch(void* const* d_in, const int* in_sizes, int n_in,
                              void* d_out, int out_size, void* d_ws, size_t ws_size,
                              hipStream_t stream) {
  (void)in_sizes; (void)n_in; (void)out_size; (void)ws_size;

  const int*   ids     = (const int*)d_in[0];
  const int*   slen    = (const int*)d_in[1];
  const float* ch_emb  = (const float*)d_in[2];
  const float* pos_emb = (const float*)d_in[3];
  const float* Wq      = (const float*)d_in[4];
  const float* bq      = (const float*)d_in[5];
  const float* Wk      = (const float*)d_in[6];
  const float* bk      = (const float*)d_in[7];
  const float* Wv      = (const float*)d_in[8];
  const float* bv      = (const float*)d_in[9];
  const float* gamma   = (const float*)d_in[10];
  const float* beta    = (const float*)d_in[11];
  const float* W1      = (const float*)d_in[12];
  const float* b1      = (const float*)d_in[13];
  const float* W2      = (const float*)d_in[14];
  const float* b2      = (const float*)d_in[15];
  const float* Wo      = (const float*)d_in[16];
  const float* bo      = (const float*)d_in[17];
  float* out = (float*)d_out;

  // ---- workspace carve-out (256B aligned regions) ----
  char* cur = (char*)d_ws;
  auto alloc_f = [&](size_t n) {
    float* p = (float*)cur; cur += ((n * 4 + 255) / 256) * 256; return p;
  };
  auto alloc_h = [&](size_t n) {
    ushort_t* p = (ushort_t*)cur; cur += ((n * 2 + 255) / 256) * 256; return p;
  };

  const size_t R = (size_t)NTOK * DD;
  float*    x    = alloc_f(R);          // fp32 residual stream
  float*    t    = alloc_f(R);          // attention out / scratch fp32
  float*    u    = alloc_f(R);          // ffn2 out fp32
  ushort_t* xbf  = alloc_h(R);
  ushort_t* qbf  = alloc_h(R);
  ushort_t* kbf  = alloc_h(R);
  ushort_t* vbf  = alloc_h(R);
  ushort_t* f1bf = alloc_h((size_t)NTOK * MFF);
  ushort_t* wqT  = alloc_h((size_t)4 * DD * DD);   // [l][N=D][K=D] bf16
  ushort_t* wkT  = alloc_h((size_t)4 * DD * DD);
  ushort_t* wvT  = alloc_h((size_t)4 * DD * DD);
  ushort_t* w1T  = alloc_h((size_t)MFF * DD);      // [N=1024][K=256]
  ushort_t* w2T  = alloc_h((size_t)DD * MFF);      // [N=256][K=1024]
  ushort_t* woT  = alloc_h((size_t)LAB * DD);      // [N=20][K=256]

  auto convert = [&](const float* W, ushort_t* Wt, int K, int N) {
    size_t n = (size_t)K * N;
    convert_wT_kernel<<<(int)((n + 255) / 256), 256, 0, stream>>>(W, Wt, K, N);
  };
  for (int l = 0; l < 4; ++l) {
    const size_t wo = (size_t)l * DD * DD;
    convert(Wq + wo, wqT + wo, DD, DD);
    convert(Wk + wo, wkT + wo, DD, DD);
    convert(Wv + wo, wvT + wo, DD, DD);
  }
  convert(W1, w1T, DD, MFF);
  convert(W2, w2T, MFF, DD);
  convert(Wo, woT, DD, LAB);

  auto gemm = [&](const ushort_t* A, const ushort_t* Wt, const float* bias,
                  float* Cf, ushort_t* Cbf, int M, int N, int K) {
    int tiles = (M >> 4) * (N >> 6);
    gemm_bf_fast<<<(tiles + 7) / 8, 256, 0, stream>>>(A, Wt, bias, Cf, Cbf,
                                                      M, N, K);
  };

  embed_kernel<<<(int)(R / 256), 256, 0, stream>>>(ids, ch_emb, pos_emb, x, xbf);

  for (int l = 0; l < 4; ++l) {
    const size_t wo = (size_t)l * DD * DD;
    gemm(xbf, wqT + wo, bq + (size_t)l * DD, nullptr, qbf, NTOK, DD, DD);
    gemm(xbf, wkT + wo, bk + (size_t)l * DD, nullptr, kbf, NTOK, DD, DD);
    gemm(xbf, wvT + wo, bv + (size_t)l * DD, nullptr, vbf, NTOK, DD, DD);

    attention_kernel<<<dim3(SS / 16, HH, BB), 32, 0, stream>>>(qbf, kbf, vbf,
                                                               slen, t);
    add_ln_kernel<<<NTOK, DD, 0, stream>>>(t, x, xbf, gamma, beta);

    gemm(xbf,  w1T, b1, nullptr, f1bf, NTOK, MFF, DD);   // FFN (no activation)
    gemm(f1bf, w2T, b2, u, nullptr,    NTOK, DD,  MFF);
    add_ln_kernel<<<NTOK, DD, 0, stream>>>(u, x, xbf, gamma, beta);
  }

  {  // logits [NTOK, 20], guarded N
    int tiles = (NTOK >> 4) * ((LAB + 15) >> 4);
    gemm_bf_guard<<<(tiles + 7) / 8, 256, 0, stream>>>(xbf, woT, bo, out,
                                                       NTOK, LAB, DD);
  }
}
